// Attention_69028714382091
// MI455X (gfx1250) — matrix operand order repaired and checked by reading, not compiled, not run on hardware
//
#include <hip/hip_runtime.h>

// ---------------------------------------------------------------------------
// Causal MHA forward for MI455X (gfx1250, wave32, WMMA).
//   x[128,256,384] f32 -> qkv proj -> flash attention -> out proj -> f32 out
// All matmuls via v_wmma_f32_16x16x32_bf16 (f32 accumulate).
// GEMMs use 64x32 per-wave register blocking: 8 WMMAs per 12 b128 loads,
// ~2.7x less L2 operand traffic than 16x16-per-wave.
// ---------------------------------------------------------------------------

static constexpr int BB   = 128;          // batch
static constexpr int TT   = 256;          // sequence length
static constexpr int CC   = 384;          // embed dim
static constexpr int NH   = 6;            // heads
static constexpr int DH   = 64;           // head dim
static constexpr int MM   = BB * TT;      // 32768 tokens
static constexpr int QKVN = 3 * CC;       // 1152

typedef __attribute__((ext_vector_type(16))) __bf16 v16bf;
typedef __attribute__((ext_vector_type(8)))  float  v8f;

union BF16Tile {
  v16bf v;
  uint4 q[2];
};

__device__ __forceinline__ v8f vzero8() {
  v8f z;
#pragma unroll
  for (int i = 0; i < 8; ++i) z[i] = 0.0f;
  return z;
}

__device__ __forceinline__ unsigned short f32_to_bf16(float f) {
  unsigned int u = __float_as_uint(f);
  u += 0x7FFFu + ((u >> 16) & 1u);          // round to nearest even
  return (unsigned short)(u >> 16);
}

__device__ __forceinline__ v8f wmma_bf16(const BF16Tile& a, const BF16Tile& b, v8f c) {
  // (neg_a, A, neg_b, B, c_mod, C, reuse_a, reuse_b)
  return __builtin_amdgcn_wmma_f32_16x16x32_bf16(false, a.v, false, b.v,
                                                 (short)0, c, false, false);
}

// ---------------------------------------------------------------------------
// fp32 -> bf16 conversion
// ---------------------------------------------------------------------------
__global__ void cvt_f32_bf16_kernel(const float* __restrict__ src,
                                    unsigned short* __restrict__ dst, int n) {
  int i = blockIdx.x * blockDim.x + threadIdx.x;
  if (i < n) dst[i] = f32_to_bf16(src[i]);
}

// ---------------------------------------------------------------------------
// Core: one wave computes a 64(M) x 32(N) f32 macro-tile of A[MxK]*B[NxK]^T
// (both bf16, K contiguous) as a 4x2 grid of 16x16 WMMA tiles.
// A-operand layout (ISA 7.12.2): lane(l<16): m=l, K runs {0..7},{16..23};
// lane(l>=16): m=l-16, K runs {8..15},{24..31}.
// B-operand: lane l holds column n=l%16, K runs {8h..8h+7},{16+8h..23+8h}, h=l/16 (same as A).
// ---------------------------------------------------------------------------
__device__ __forceinline__ void bf16_gemm_tile64x32(
    const unsigned short* __restrict__ A,
    const unsigned short* __restrict__ Bm,
    int m0, int n0, int K, int lane, v8f acc[8]) {
  const int lo = lane & 15;
  const int hi = lane >> 4;
  const unsigned short* ap0 = A  + (size_t)(m0 + lo) * K;
  const unsigned short* bp0 = Bm + (size_t)(n0 + lo) * K + 8 * hi;
  const int ao = 8 * hi;
#pragma unroll
  for (int i = 0; i < 8; ++i) acc[i] = vzero8();
  for (int k0 = 0; k0 < K; k0 += 32) {
    BF16Tile a[4], b[2];
#pragma unroll
    for (int i = 0; i < 4; ++i) {
      const unsigned short* ap = ap0 + (size_t)(16 * i) * K + k0;
      a[i].q[0] = *(const uint4*)(ap + ao);
      a[i].q[1] = *(const uint4*)(ap + 16 + ao);
    }
#pragma unroll
    for (int j = 0; j < 2; ++j) {
      const unsigned short* bp = bp0 + (size_t)(16 * j) * K + k0;
      b[j].q[0] = *(const uint4*)(bp);
      b[j].q[1] = *(const uint4*)(bp + 16);
    }
    __builtin_prefetch((const void*)(ap0 + k0 + 128), 0, 3);  // global_prefetch_b8
    __builtin_prefetch((const void*)(bp0 + k0 + 128), 0, 3);
#pragma unroll
    for (int i = 0; i < 4; ++i)
#pragma unroll
      for (int j = 0; j < 2; ++j)
        acc[i * 2 + j] = wmma_bf16(a[i], b[j], acc[i * 2 + j]);
  }
}

// ---------------------------------------------------------------------------
// QKV projection: qkv[m,o] = x[m,:] . W_qkv[o,:] + b_qkv[o]
// Epilogue scatters to Q,K as [b,h,t,d] bf16 and V transposed [b,h,d,t] bf16.
// Grid: (MM/64)*(QKVN/32) = 18432 wave-tiles, 8 waves / 256-thread block.
// ---------------------------------------------------------------------------
__global__ void __launch_bounds__(256)
qkv_gemm_kernel(const unsigned short* __restrict__ xb,
                const unsigned short* __restrict__ wb,
                const float* __restrict__ bias,
                unsigned short* __restrict__ qb,
                unsigned short* __restrict__ kb,
                unsigned short* __restrict__ vtb) {
  const int wid  = blockIdx.x * 8 + (threadIdx.x >> 5);
  const int lane = threadIdx.x & 31;
  const int lo = lane & 15, hi = lane >> 4;
  const int NT = QKVN / 32;                  // 36
  const int m0 = (wid / NT) * 64, n0 = (wid % NT) * 32;

  v8f acc[8];
  bf16_gemm_tile64x32(xb, wb, m0, n0, CC, lane, acc);

#pragma unroll
  for (int j = 0; j < 2; ++j) {
    const int col = n0 + 16 * j + lo;        // 0..1151
    const float bv = bias[col];
    const int part = col / CC;               // 0=q 1=k 2=v
    const int c = col % CC;
    const int h = c >> 6, d = c & 63;
#pragma unroll
    for (int i = 0; i < 4; ++i) {
#pragma unroll
      for (int r = 0; r < 8; ++r) {
        const int row = m0 + 16 * i + r + 8 * hi;   // token index
        const int b = row >> 8, t = row & 255;
        const unsigned short bf = f32_to_bf16(acc[i * 2 + j][r] + bv);
        const size_t bh = (size_t)(b * NH + h);
        if (part == 0)      qb [(bh * TT + t) * DH + d] = bf;
        else if (part == 1) kb [(bh * TT + t) * DH + d] = bf;
        else                vtb[(bh * DH + d) * TT + t] = bf;   // transposed V
      }
    }
  }
}

// ---------------------------------------------------------------------------
// Flash attention: block = 4 waves x 32 lanes; wave owns a 16-row query tile.
// blockIdx.x = bh*4 + (64-row group). Streams 32-key blocks with online
// softmax; P goes through wave-private LDS to transpose C-layout -> A-layout.
// ---------------------------------------------------------------------------
__global__ void __launch_bounds__(128)
attn_flash_kernel(const unsigned short* __restrict__ qb,
                  const unsigned short* __restrict__ kbuf,
                  const unsigned short* __restrict__ vtb,
                  unsigned short* __restrict__ attnb) {
  __shared__ __align__(16) unsigned short Pbuf[4][16 * 32];

  const int w    = threadIdx.x >> 5;
  const int lane = threadIdx.x & 31;
  const int lo = lane & 15, hi = lane >> 4;
  const int bh = blockIdx.x >> 2;
  const int q0 = (blockIdx.x & 3) * 64 + w * 16;
  const int b  = bh / NH, h = bh % NH;

  // Q tile as two A operands (Dh chunks 0..31, 32..63), resident for the loop.
  const unsigned short* qrow = qb + ((size_t)bh * TT + q0 + lo) * DH;
  BF16Tile qa0, qa1;
  qa0.q[0] = *(const uint4*)(qrow + 16 * hi);
  qa0.q[1] = *(const uint4*)(qrow + 8 + 16 * hi);
  qa1.q[0] = *(const uint4*)(qrow + 32 + 16 * hi);
  qa1.q[1] = *(const uint4*)(qrow + 40 + 16 * hi);

  v8f oacc[4];
#pragma unroll
  for (int n = 0; n < 4; ++n) oacc[n] = vzero8();
  float mrow[8], lrow[8];
#pragma unroll
  for (int r = 0; r < 8; ++r) { mrow[r] = -3.0e38f; lrow[r] = 0.0f; }

  const float scale = 0.125f;                 // 1/sqrt(64)
  const int nblk = (q0 + 47) >> 5;            // causal key-block bound

  for (int jb = 0; jb < nblk; ++jb) {
    const int kb0 = jb * 32;

    // ---- S = Q K^T for 32 keys as two 16x16 subtiles ----
    const unsigned short* kr = kbuf + ((size_t)bh * TT + kb0 + lo) * DH + 16 * hi;
    BF16Tile t0, t1;
    v8f s0 = vzero8(), s1 = vzero8();
    t0.q[0] = *(const uint4*)(kr);       t0.q[1] = *(const uint4*)(kr + 8);
    t1.q[0] = *(const uint4*)(kr + 32);  t1.q[1] = *(const uint4*)(kr + 40);
    s0 = wmma_bf16(qa0, t0, s0);
    s0 = wmma_bf16(qa1, t1, s0);
    const unsigned short* kr2 = kr + 16 * DH;   // keys +16
    t0.q[0] = *(const uint4*)(kr2);      t0.q[1] = *(const uint4*)(kr2 + 8);
    t1.q[0] = *(const uint4*)(kr2 + 32); t1.q[1] = *(const uint4*)(kr2 + 40);
    s1 = wmma_bf16(qa0, t0, s1);
    s1 = wmma_bf16(qa1, t1, s1);

    // ---- online softmax over this key block ----
    const int col0 = kb0 + lo;
    const int col1 = col0 + 16;
    float p0[8], p1[8];
#pragma unroll
    for (int r = 0; r < 8; ++r) {
      const int row = q0 + r + 8 * hi;
      float v0 = s0[r] * scale; if (col0 > row) v0 = -3.0e38f;
      float v1 = s1[r] * scale; if (col1 > row) v1 = -3.0e38f;
      float mx = fmaxf(v0, v1);
#pragma unroll
      for (int msk = 1; msk < 16; msk <<= 1) mx = fmaxf(mx, __shfl_xor(mx, msk, 32));
      const float mnew  = fmaxf(mrow[r], mx);
      const float alpha = __expf(mrow[r] - mnew);
      const float e0 = __expf(v0 - mnew);
      const float e1 = __expf(v1 - mnew);
      float sum = e0 + e1;
#pragma unroll
      for (int msk = 1; msk < 16; msk <<= 1) sum += __shfl_xor(sum, msk, 32);
      lrow[r] = lrow[r] * alpha + sum;
      mrow[r] = mnew;
#pragma unroll
      for (int n = 0; n < 4; ++n) oacc[n][r] *= alpha;
      p0[r] = e0; p1[r] = e1;
    }

    // ---- P: C-layout -> bf16 A-layout via wave-private LDS ----
#pragma unroll
    for (int r = 0; r < 8; ++r) {
      const int prow = r + 8 * hi;
      Pbuf[w][prow * 32 + lo]      = f32_to_bf16(p0[r]);
      Pbuf[w][prow * 32 + 16 + lo] = f32_to_bf16(p1[r]);
    }
    BF16Tile pa;
    const unsigned short* pr = &Pbuf[w][lo * 32];
    pa.q[0] = *(const uint4*)(pr + 16 * hi);
    pa.q[1] = *(const uint4*)(pr + 8 + 16 * hi);

    // ---- O += P V  (V transposed layout -> contiguous B-operand loads) ----
#pragma unroll
    for (int n = 0; n < 4; ++n) {
      const unsigned short* vp =
          vtb + ((size_t)bh * DH + 16 * n + lo) * TT + kb0 + 16 * hi;
      BF16Tile vb;
      vb.q[0] = *(const uint4*)(vp);
      vb.q[1] = *(const uint4*)(vp + 8);
      oacc[n] = wmma_bf16(pa, vb, oacc[n]);
    }
  }

  // ---- finalize: O /= l, store bf16 into [B,T,C] attn buffer ----
#pragma unroll
  for (int r = 0; r < 8; ++r) {
    const float inv = 1.0f / lrow[r];
    const int trow = q0 + r + 8 * hi;
    const size_t base = ((size_t)b * TT + trow) * CC + h * DH + lo;
#pragma unroll
    for (int n = 0; n < 4; ++n)
      attnb[base + 16 * n] = f32_to_bf16(oacc[n][r] * inv);
  }
}

// ---------------------------------------------------------------------------
// Output projection: out[m,o] = attn[m,:] . W_proj[o,:] + b_proj[o]  (f32 out)
// Grid: (MM/64)*(CC/32) = 6144 wave-tiles, 8 waves / 256-thread block.
// ---------------------------------------------------------------------------
__global__ void __launch_bounds__(256)
proj_gemm_kernel(const unsigned short* __restrict__ ab,
                 const unsigned short* __restrict__ wb,
                 const float* __restrict__ bias,
                 float* __restrict__ out) {
  const int wid  = blockIdx.x * 8 + (threadIdx.x >> 5);
  const int lane = threadIdx.x & 31;
  const int lo = lane & 15, hi = lane >> 4;
  const int NT = CC / 32;                    // 12
  const int m0 = (wid / NT) * 64, n0 = (wid % NT) * 32;

  v8f acc[8];
  bf16_gemm_tile64x32(ab, wb, m0, n0, CC, lane, acc);

#pragma unroll
  for (int j = 0; j < 2; ++j) {
    const int col = n0 + 16 * j + lo;
    const float bv = bias[col];
#pragma unroll
    for (int i = 0; i < 4; ++i) {
#pragma unroll
      for (int r = 0; r < 8; ++r) {
        const int row = m0 + 16 * i + r + 8 * hi;
        out[(size_t)row * CC + col] = acc[i * 2 + j][r] + bv;
      }
    }
  }
}

// ---------------------------------------------------------------------------
extern "C" void kernel_launch(void* const* d_in, const int* in_sizes, int n_in,
                              void* d_out, int out_size, void* d_ws, size_t ws_size,
                              hipStream_t stream) {
  (void)in_sizes; (void)n_in; (void)out_size; (void)ws_size;
  const float* x     = (const float*)d_in[0];
  const float* Wqkv  = (const float*)d_in[1];
  const float* bqkv  = (const float*)d_in[2];
  const float* Wproj = (const float*)d_in[3];
  const float* bproj = (const float*)d_in[4];
  float* out = (float*)d_out;

  char* ws = (char*)d_ws;
  size_t off = 0;
  auto take = [&](size_t bytes) -> unsigned short* {
    unsigned short* p = (unsigned short*)(ws + off);
    off += (bytes + 255) & ~(size_t)255;
    return p;
  };
  unsigned short* xb  = take((size_t)MM * CC * 2);      // x bf16 (later: attn out)
  unsigned short* wqb = take((size_t)QKVN * CC * 2);    // W_qkv bf16
  unsigned short* wpb = take((size_t)CC * CC * 2);      // W_proj bf16
  unsigned short* qb  = take((size_t)MM * CC * 2);      // Q [b,h,t,d]
  unsigned short* kb  = take((size_t)MM * CC * 2);      // K [b,h,t,d]
  unsigned short* vtb = take((size_t)MM * CC * 2);      // V [b,h,d,t]

  cvt_f32_bf16_kernel<<<(MM * CC + 255) / 256, 256, 0, stream>>>(x, xb, MM * CC);
  cvt_f32_bf16_kernel<<<(QKVN * CC + 255) / 256, 256, 0, stream>>>(Wqkv, wqb, QKVN * CC);
  cvt_f32_bf16_kernel<<<(CC * CC + 255) / 256, 256, 0, stream>>>(Wproj, wpb, CC * CC);

  // (32768/64)*(1152/32) = 18432 wave tiles / 8 waves per block
  qkv_gemm_kernel<<<(MM / 64) * (QKVN / 32) / 8, 256, 0, stream>>>(
      xb, wqb, bqkv, qb, kb, vtb);

  unsigned short* attnb = xb;  // reuse: x bf16 fully consumed by qkv_gemm
  attn_flash_kernel<<<BB * NH * (TT / 64), 128, 0, stream>>>(qb, kb, vtb, attnb);

  // (32768/64)*(384/32) = 6144 wave tiles / 8 waves per block
  proj_gemm_kernel<<<(MM / 64) * (CC / 32) / 8, 256, 0, stream>>>(
      attnb, wpb, bproj, out);
}
